// DifferenceEncoder_33174327394777
// MI455X (gfx1250) — matrix operand order repaired and checked by reading, not compiled, not run on hardware
//
#include <hip/hip_runtime.h>
#include <hip/hip_bf16.h>

typedef __attribute__((ext_vector_type(16))) _Float16 v16h;
typedef __attribute__((ext_vector_type(8)))  _Float16 v8h;
typedef __attribute__((ext_vector_type(8)))  float    v8f;

// ---------------------------------------------------------------------------
// WMMA fragment helpers (gfx1250, wave32, 16x16x32 f16 -> f32)
// A: 16(M) x 32(K) from row-major (M x K) f16, rows 16B-aligned (ld % 8 == 0)
//   lane = 16*half + m ; VGPR0-3 hold K = lo..lo+7 (lo = half?8:0),
//   VGPR4-7 hold K = 16+lo..16+lo+7  (per ISA A-matrix table)
// B: 32(K) x 16(N) with storage row-major (N x K):
//   lane = 16*half + n ; same K order as A (lo..lo+7, 16+lo..16+lo+7)
// ---------------------------------------------------------------------------
__device__ __forceinline__ v16h load_a16(const _Float16* __restrict__ base,
                                         int ld, int row0, int k0) {
  int lane = threadIdx.x & 31;
  int r  = lane & 15;
  int lo = (lane & 16) ? 8 : 0;
  const _Float16* p = base + (size_t)(row0 + r) * ld + k0 + lo;
  v8h u0 = *(const v8h*)(p);
  v8h u1 = *(const v8h*)(p + 16);
  v16h f;
#pragma unroll
  for (int i = 0; i < 8; ++i) { f[i] = u0[i]; f[i + 8] = u1[i]; }
  return f;
}

__device__ __forceinline__ v16h load_b16(const _Float16* __restrict__ base,
                                         int ld, int n0, int k0) {
  int lane = threadIdx.x & 31;
  int n   = lane & 15;
  int off = (lane & 16) ? 8 : 0;
  const _Float16* p = base + (size_t)(n0 + n) * ld + k0 + off;
  v8h u0 = *(const v8h*)(p);
  v8h u1 = *(const v8h*)(p + 16);
  v16h f;
#pragma unroll
  for (int i = 0; i < 8; ++i) { f[i] = u0[i]; f[i + 8] = u1[i]; }
  return f;
}

__device__ __forceinline__ v8f wmma16(v16h a, v16h b, v8f c) {
  return __builtin_amdgcn_wmma_f32_16x16x32_f16(false, a, false, b,
                                                (short)0, c, false, false);
}

// ---------------------------------------------------------------------------
// Generic GEMM: out(MxN) = A(MxK,f16,row-major) @ W(NxK,f16,row-major)^T + bias
// optionally writes f32 and/or scaled f16 outputs. 8 waves/block -> 32x64 tile.
// ---------------------------------------------------------------------------
__global__ void gemm_wmma_kernel(const _Float16* __restrict__ A,
                                 const _Float16* __restrict__ Wt,
                                 const float* __restrict__ bias,
                                 float* __restrict__ outF,
                                 _Float16* __restrict__ outH,
                                 float hscale, int M, int N, int K) {
  int wave = threadIdx.x >> 5;
  int lane = threadIdx.x & 31;
  int m0 = blockIdx.x * 32 + (wave & 1) * 16;
  int n0 = blockIdx.y * 64 + (wave >> 1) * 16;
  if (m0 >= M || n0 >= N) return;   // wave-uniform
  v8f acc = {};
  for (int k0 = 0; k0 < K; k0 += 32) {
    v16h a = load_a16(A, K, m0, k0);
    v16h b = load_b16(Wt, K, n0, k0);
    acc = wmma16(a, b, acc);
  }
  int col = n0 + (lane & 15);
  float bs = bias ? bias[col] : 0.0f;
  int rbase = (lane & 16) ? 8 : 0;
#pragma unroll
  for (int r = 0; r < 8; ++r) {
    int row = m0 + rbase + r;
    float v = acc[r] + bs;
    size_t idx = (size_t)row * N + col;
    if (outF) outF[idx] = v;
    if (outH) outH[idx] = (_Float16)(v * hscale);
  }
}

// ---------------------------------------------------------------------------
// Flash attention with analytic Manhattan-decay mask. One wave per
// (16-query block, head). head_dim = 32 always. Q/K/V are (L x C) f16,
// K pre-scaled by d^-0.5. Output f32 (L x C).
// ---------------------------------------------------------------------------
__global__ void flash_attn_kernel(const _Float16* __restrict__ Q,
                                  const _Float16* __restrict__ Kh,
                                  const _Float16* __restrict__ Vh,
                                  float* __restrict__ O,
                                  int L, int C, int Wdim, int nheads,
                                  float init_v, float h_range) {
  __shared__ _Float16 Pt[16 * 32];
  __shared__ _Float16 Vt[32 * 32];
  int lane = threadIdx.x & 31;
  int q0   = blockIdx.x * 16;
  int head = blockIdx.y;
  int ho   = head * 32;
  float decay = logf(1.0f - exp2f(-(init_v + h_range * (float)head / (float)nheads)));

  v16h qa = load_a16(Q, C, q0, ho);
  int rbase = (lane & 16) ? 8 : 0;
  int colk  = lane & 15;

  float m[8], lsum[8];
#pragma unroll
  for (int r = 0; r < 8; ++r) { m[r] = -1e30f; lsum[r] = 0.0f; }
  v8f o0 = {}, o1 = {};

  for (int j0 = 0; j0 < L; j0 += 32) {
    // stage V tile (32 keys x 32 dims) into LDS, row-major
    {
      const _Float16* src = Vh + (size_t)(j0 + lane) * C + ho;
      v8h a0 = *(const v8h*)(src);
      v8h a1 = *(const v8h*)(src + 8);
      v8h a2 = *(const v8h*)(src + 16);
      v8h a3 = *(const v8h*)(src + 24);
      _Float16* d = &Vt[lane * 32];
      *(v8h*)(d) = a0; *(v8h*)(d + 8) = a1; *(v8h*)(d + 16) = a2; *(v8h*)(d + 24) = a3;
    }
    __syncthreads();

    v16h b0 = load_b16(Kh, C, j0,      ho);
    v16h b1 = load_b16(Kh, C, j0 + 16, ho);
    v8f s0 = {}, s1 = {};
    s0 = wmma16(qa, b0, s0);
    s1 = wmma16(qa, b1, s1);

    int kt0 = j0 + colk, kt1 = j0 + 16 + colk;
    int k0h = kt0 / Wdim, k0w = kt0 % Wdim;
    int k1h = kt1 / Wdim, k1w = kt1 % Wdim;

    float p0[8], p1[8];
#pragma unroll
    for (int r = 0; r < 8; ++r) {
      int qt = q0 + rbase + r;
      int qh = qt / Wdim, qw = qt % Wdim;
      int d0 = (qh > k0h ? qh - k0h : k0h - qh) + (qw > k0w ? qw - k0w : k0w - qw);
      int d1 = (qh > k1h ? qh - k1h : k1h - qh) + (qw > k1w ? qw - k1w : k1w - qw);
      float v0 = s0[r] + decay * (float)d0;
      float v1 = s1[r] + decay * (float)d1;
      float mr = fmaxf(v0, v1);
#pragma unroll
      for (int dd = 1; dd < 16; dd <<= 1) mr = fmaxf(mr, __shfl_xor(mr, dd, 32));
      float mnew = fmaxf(m[r], mr);
      float alpha = __expf(m[r] - mnew);
      p0[r] = __expf(v0 - mnew);
      p1[r] = __expf(v1 - mnew);
      float rs = p0[r] + p1[r];
#pragma unroll
      for (int dd = 1; dd < 16; dd <<= 1) rs += __shfl_xor(rs, dd, 32);
      lsum[r] = lsum[r] * alpha + rs;
      o0[r] *= alpha;
      o1[r] *= alpha;
      m[r] = mnew;
    }
    // write P (16 x 32) to LDS in row-major f16
#pragma unroll
    for (int r = 0; r < 8; ++r) {
      int row = rbase + r;
      Pt[row * 32 + colk]      = (_Float16)p0[r];
      Pt[row * 32 + 16 + colk] = (_Float16)p1[r];
    }
    __syncthreads();

    // A fragment of P from LDS
    v16h pa;
    {
      int rr = lane & 15;
      int lo = (lane & 16) ? 16 : 0;
#pragma unroll
      for (int i = 0; i < 8; ++i) {
        pa[i]     = Pt[rr * 32 + lo + i];
        pa[i + 8] = Pt[rr * 32 + 8 + lo + i];
      }
    }
    // B fragments of V from LDS (B[k][n] = Vt[k][n0h + n])
    v16h bv0, bv1;
    {
      int n    = lane & 15;
      int koff = (lane & 16) ? 16 : 0;
#pragma unroll
      for (int i = 0; i < 16; ++i) {
        bv0[i] = Vt[(koff + i) * 32 + n];
        bv1[i] = Vt[(koff + i) * 32 + 16 + n];
      }
    }
    o0 = wmma16(pa, bv0, o0);
    o1 = wmma16(pa, bv1, o1);
    __syncthreads();
  }
#pragma unroll
  for (int r = 0; r < 8; ++r) {
    float inv = 1.0f / lsum[r];
    int row = q0 + rbase + r;
    O[(size_t)row * C + ho + colk]      = o0[r] * inv;
    O[(size_t)row * C + ho + 16 + colk] = o1[r] * inv;
  }
}

// ---------------------------------------------------------------------------
// Per-row LayerNorm over C (contiguous). One wave per row (wave32 shuffles).
// ---------------------------------------------------------------------------
__global__ void ln_rows_kernel(const float* __restrict__ src,
                               const float* __restrict__ gamma,
                               const float* __restrict__ beta,
                               float* __restrict__ outF,
                               _Float16* __restrict__ outH,
                               int L, int C) {
  int wave = threadIdx.x >> 5, lane = threadIdx.x & 31;
  int row = blockIdx.x * (blockDim.x >> 5) + wave;
  if (row >= L) return;
  const float* p = src + (size_t)row * C;
  float s = 0.f, s2 = 0.f;
  for (int c = lane; c < C; c += 32) { float v = p[c]; s += v; s2 += v * v; }
#pragma unroll
  for (int d = 1; d < 32; d <<= 1) { s += __shfl_xor(s, d, 32); s2 += __shfl_xor(s2, d, 32); }
  float mean = s / (float)C;
  float var  = s2 / (float)C - mean * mean;
  float inv  = rsqrtf(var + 1e-6f);
  for (int c = lane; c < C; c += 32) {
    float v = (p[c] - mean) * inv * gamma[c] + beta[c];
    if (outF) outF[(size_t)row * C + c] = v;
    if (outH) outH[(size_t)row * C + c] = (_Float16)v;
  }
}

// ---------------------------------------------------------------------------
// Depthwise KxK conv on (L x C) NHWC f32, pad, + bias.
// ---------------------------------------------------------------------------
__global__ void dwconv_kernel(const float* __restrict__ src,
                              const float* __restrict__ wgt,
                              const float* __restrict__ bias,
                              float* __restrict__ dst,
                              int H, int W, int C, int ksz, int pad) {
  int idx = blockIdx.x * blockDim.x + threadIdx.x;
  int total = H * W * C;
  if (idx >= total) return;
  int c = idx % C;
  int l = idx / C;
  int w = l % W, h = l / W;
  float acc = bias[c];
  for (int kh = 0; kh < ksz; ++kh) {
    int hh = h + kh - pad;
    if (hh < 0 || hh >= H) continue;
    for (int kw = 0; kw < ksz; ++kw) {
      int ww = w + kw - pad;
      if (ww < 0 || ww >= W) continue;
      acc += src[((size_t)(hh * W + ww)) * C + c] * wgt[c * ksz * ksz + kh * ksz + kw];
    }
  }
  dst[idx] = acc;
}

// ---------------------------------------------------------------------------
// Small elementwise / layout kernels
// ---------------------------------------------------------------------------
__global__ void f32_to_f16_kernel(const float* __restrict__ s,
                                  _Float16* __restrict__ d, float scale, int n) {
  int i = blockIdx.x * blockDim.x + threadIdx.x;
  if (i < n) d[i] = (_Float16)(s[i] * scale);
}

// concat(a,b) along channels; NCHW f32 -> token-major (L x 2D) f16
__global__ void concat_ab_kernel(const float* __restrict__ a,
                                 const float* __restrict__ b,
                                 _Float16* __restrict__ dst, int L, int D) {
  int idx = blockIdx.x * blockDim.x + threadIdx.x;
  int K2 = 2 * D;
  if (idx >= L * K2) return;
  int c = idx % K2, l = idx / K2;
  float v = (c < D) ? a[(size_t)c * L + l] : b[(size_t)(c - D) * L + l];
  dst[idx] = (_Float16)v;
}

// concat [x, att] channels (each L x C f32) -> (L x 2C) f16
__global__ void concat_xa_kernel(const float* __restrict__ x,
                                 const float* __restrict__ att,
                                 _Float16* __restrict__ dst, int L, int C) {
  int idx = blockIdx.x * blockDim.x + threadIdx.x;
  int D = 2 * C;
  if (idx >= L * D) return;
  int c = idx % D, l = idx / D;
  float v = (c < C) ? x[(size_t)l * C + c] : att[(size_t)l * C + (c - C)];
  dst[idx] = (_Float16)v;
}

__global__ void add_gelu_kernel(float* __restrict__ x,
                                const float* __restrict__ t, int n) {
  int i = blockIdx.x * blockDim.x + threadIdx.x;
  if (i >= n) return;
  float v = t[i];
  x[i] += 0.5f * v * (1.0f + erff(v * 0.70710678118654752f));
}

__global__ void add2h_kernel(const float* __restrict__ a,
                             const float* __restrict__ b,
                             _Float16* __restrict__ d, int n) {
  int i = blockIdx.x * blockDim.x + threadIdx.x;
  if (i < n) d[i] = (_Float16)(a[i] + b[i]);
}

// im2col for 3x3 pad 1 on (L x D) f16 -> (L x 9D) f16, K index = t*D + c
__global__ void im2col_kernel(const _Float16* __restrict__ src,
                              _Float16* __restrict__ col,
                              int H, int W, int D) {
  int idx = blockIdx.x * blockDim.x + threadIdx.x;
  int total = H * W * 9 * D;
  if (idx >= total) return;
  int c = idx % D;
  int t = (idx / D) % 9;
  int l = idx / (9 * D);
  int h = l / W, w = l % W;
  int hh = h + t / 3 - 1, ww = w + t % 3 - 1;
  _Float16 v = (_Float16)0.0f;
  if (hh >= 0 && hh < H && ww >= 0 && ww < W)
    v = src[(size_t)(hh * W + ww) * D + c];
  col[idx] = v;
}

// proj weight (256, D, 3, 3) f32 -> Wt (256 x 9D) f16 with K index = t*D + c
__global__ void proj_w_kernel(const float* __restrict__ w,
                              _Float16* __restrict__ dst, int D) {
  int idx = blockIdx.x * blockDim.x + threadIdx.x;
  int total = 256 * D * 9;
  if (idx >= total) return;
  int t = idx % 9;
  int c = (idx / 9) % D;
  int n = idx / (9 * D);
  dst[(size_t)n * 9 * D + t * D + c] = (_Float16)w[idx];
}

// fuse: out NCHW(1,256,64,64) = s0 + up2(s1) + up4(s2) + up8(s3);
// scale buffers are token-major (L_s x 256)
__global__ void fuse_kernel(const float* __restrict__ s0,
                            const float* __restrict__ s1,
                            const float* __restrict__ s2,
                            const float* __restrict__ s3,
                            float* __restrict__ out) {
  int idx = blockIdx.x * blockDim.x + threadIdx.x;
  if (idx >= 256 * 64 * 64) return;
  int w = idx % 64;
  int h = (idx / 64) % 64;
  int c = idx / 4096;
  float v = s0[(size_t)(h * 64 + w) * 256 + c]
          + s1[(size_t)((h >> 1) * 32 + (w >> 1)) * 256 + c]
          + s2[(size_t)((h >> 2) * 16 + (w >> 2)) * 256 + c]
          + s3[(size_t)((h >> 3) * 8  + (w >> 3)) * 256 + c];
  out[idx] = v;
}

// ---------------------------------------------------------------------------
static inline dim3 grd(int n) { return dim3((n + 255) / 256); }

extern "C" void kernel_launch(void* const* d_in, const int* in_sizes, int n_in,
                              void* d_out, int out_size, void* d_ws, size_t ws_size,
                              hipStream_t stream) {
  (void)in_sizes; (void)n_in; (void)out_size; (void)ws_size;
  const int DIMS_[4]  = {256, 512, 1024, 2048};
  const int HEADS_[4] = {4, 8, 16, 32};
  const int HS_[4]    = {64, 32, 16, 8};

  const size_t MB = 1ull << 20;
  char* ws = (char*)d_ws;
  _Float16* W16   = (_Float16*)(ws + 0);        // 12 MB  weight f16 staging
  _Float16* BIG   = (_Float16*)(ws + 12 * MB);  // 20 MB  concat-input / im2col
  float*    XF    = (float*)(ws + 32 * MB);     // 2 MB   x (L x C)
  float*    YF    = (float*)(ws + 34 * MB);     // 2 MB   dwconv out
  float*    TF    = (float*)(ws + 36 * MB);     // 2 MB   gemm temp
  float*    VF    = (float*)(ws + 38 * MB);     // 2 MB   v f32
  float*    AF    = (float*)(ws + 40 * MB);     // 2 MB   flash out
  float*    LF    = (float*)(ws + 42 * MB);     // 2 MB   lepe out
  _Float16* H0    = (_Float16*)(ws + 44 * MB);  // ln(local) f16
  _Float16* H1    = (_Float16*)(ws + 45 * MB);  // xn f16
  _Float16* H2    = (_Float16*)(ws + 46 * MB);  // q f16
  _Float16* H3    = (_Float16*)(ws + 47 * MB);  // k f16 (pre-scaled)
  _Float16* H4    = (_Float16*)(ws + 48 * MB);  // v f16
  _Float16* H5    = (_Float16*)(ws + 49 * MB);  // att+lepe f16
  _Float16* CAT16 = (_Float16*)(ws + 50 * MB);  // 2 MB (L x D)
  float*    FF    = (float*)(ws + 52 * MB);     // 4 MB (L x 256)
  float* OUTS[4] = { (float*)(ws + 56 * MB), (float*)(ws + 60 * MB),
                     (float*)(ws + 61 * MB), (float*)(ws + 62 * MB) };

  const float kscale = 0.17677669529663688f;  // 32^-0.5

  for (int s = 0; s < 4; ++s) {
    int D = DIMS_[s], C = D / 2, H = HS_[s], W = H, L = H * W, NH = HEADS_[s];
    const float* a = (const float*)d_in[2 * s];
    const float* b = (const float*)d_in[2 * s + 1];
    const float* P[23];
    for (int j = 0; j < 23; ++j) P[j] = (const float*)d_in[8 + s * 23 + j];
    // 0 in_w 1 in_b 2 dw_w 3 dw_b 4 ln_w 5 ln_b 6 pw_w 7 pw_b 8 nii_w 9 nii_b
    // 10 q_w 11 q_b 12 k_w 13 k_b 14 v_w 15 v_b 16 lepe_w 17 lepe_b
    // 18 out_w 19 out_b 20 proj_w 21 ni_w 22 ni_b

    // ---- 1x1 input conv: x = concat(a,b) @ in_w^T + in_b ----
    concat_ab_kernel<<<grd(L * 2 * D), 256, 0, stream>>>(a, b, BIG, L, D);
    f32_to_f16_kernel<<<grd(C * 2 * D), 256, 0, stream>>>(P[0], W16, 1.f, C * 2 * D);
    gemm_wmma_kernel<<<dim3(L / 32, C / 64), 256, 0, stream>>>(
        BIG, W16, P[1], XF, nullptr, 1.f, L, C, 2 * D);

    // ---- local block: dw3x3 -> LN -> linear -> GELU -> residual ----
    dwconv_kernel<<<grd(L * C), 256, 0, stream>>>(XF, P[2], P[3], YF, H, W, C, 3, 1);
    ln_rows_kernel<<<(L + 7) / 8, 256, 0, stream>>>(YF, P[4], P[5], nullptr, H0, L, C);
    f32_to_f16_kernel<<<grd(C * C), 256, 0, stream>>>(P[6], W16, 1.f, C * C);
    gemm_wmma_kernel<<<dim3(L / 32, C / 64), 256, 0, stream>>>(
        H0, W16, P[7], TF, nullptr, 1.f, L, C, C);
    add_gelu_kernel<<<grd(L * C), 256, 0, stream>>>(XF, TF, L * C);

    // ---- pre-attention channel LN ----
    ln_rows_kernel<<<(L + 7) / 8, 256, 0, stream>>>(XF, P[8], P[9], nullptr, H1, L, C);

    // ---- q, k (scaled), v projections ----
    f32_to_f16_kernel<<<grd(C * C), 256, 0, stream>>>(P[10], W16, 1.f, C * C);
    gemm_wmma_kernel<<<dim3(L / 32, C / 64), 256, 0, stream>>>(
        H1, W16, P[11], nullptr, H2, 1.f, L, C, C);
    f32_to_f16_kernel<<<grd(C * C), 256, 0, stream>>>(P[12], W16, 1.f, C * C);
    gemm_wmma_kernel<<<dim3(L / 32, C / 64), 256, 0, stream>>>(
        H1, W16, P[13], nullptr, H3, kscale, L, C, C);
    f32_to_f16_kernel<<<grd(C * C), 256, 0, stream>>>(P[14], W16, 1.f, C * C);
    gemm_wmma_kernel<<<dim3(L / 32, C / 64), 256, 0, stream>>>(
        H1, W16, P[15], VF, H4, 1.f, L, C, C);

    // ---- flash attention with analytic decay mask ----
    flash_attn_kernel<<<dim3(L / 16, NH), 32, 0, stream>>>(
        H2, H3, H4, AF, L, C, W, NH, 2.0f, 3.0f);

    // ---- LePE (dw5x5 on v) + out projection ----
    dwconv_kernel<<<grd(L * C), 256, 0, stream>>>(VF, P[16], P[17], LF, H, W, C, 5, 2);
    add2h_kernel<<<grd(L * C), 256, 0, stream>>>(AF, LF, H5, L * C);
    f32_to_f16_kernel<<<grd(C * C), 256, 0, stream>>>(P[18], W16, 1.f, C * C);
    gemm_wmma_kernel<<<dim3(L / 32, C / 64), 256, 0, stream>>>(
        H5, W16, P[19], TF, nullptr, 1.f, L, C, C);

    // ---- concat [x, att] -> 3x3 proj conv (im2col GEMM, K = 9D) -> LN ----
    concat_xa_kernel<<<grd(L * D), 256, 0, stream>>>(XF, TF, CAT16, L, C);
    im2col_kernel<<<grd(L * 9 * D), 256, 0, stream>>>(CAT16, BIG, H, W, D);
    proj_w_kernel<<<grd(256 * D * 9), 256, 0, stream>>>(P[20], W16, D);
    gemm_wmma_kernel<<<dim3(L / 32, 256 / 64), 256, 0, stream>>>(
        BIG, W16, nullptr, FF, nullptr, 1.f, L, 256, 9 * D);
    ln_rows_kernel<<<(L + 7) / 8, 256, 0, stream>>>(FF, P[21], P[22], OUTS[s], nullptr, L, 256);
  }

  // ---- nearest-upsample fuse to (1,256,64,64) NCHW ----
  fuse_kernel<<<grd(256 * 64 * 64), 256, 0, stream>>>(
      OUTS[0], OUTS[1], OUTS[2], OUTS[3], (float*)d_out);
}
